// PatchQwen3NextGatedDeltaNet_81106162417887
// MI455X (gfx1250) — compile-verified, matrix operand order checked by reading, not run on hardware
//
#include <hip/hip_runtime.h>
#include <hip/hip_bf16.h>

// ---------------------------------------------------------------------------
// Problem constants (from the reference)
// ---------------------------------------------------------------------------
constexpr int B_ = 2, T_ = 2048, H_ = 2048;
constexpr int NUM_V = 32, NUM_K = 16, DK = 128, DV = 128, R_ = 2;
constexpr int KEY_DIM  = NUM_K * DK;            // 2048
constexpr int VAL_DIM  = NUM_V * DV;            // 4096
constexpr int CONV_DIM = 2 * KEY_DIM + VAL_DIM; // 8192
constexpr int QKVZ_DIM = 2 * KEY_DIM + 2 * VAL_DIM; // 12288
constexpr int GRP = 2 * DK + 2 * R_ * DV;       // 768 per key-head group
constexpr int KCONV = 4;
constexpr int BT = B_ * T_;                     // 4096 rows

typedef __bf16 bf16_t;
typedef bf16_t v16bf __attribute__((ext_vector_type(16)));
typedef bf16_t v8bf  __attribute__((ext_vector_type(8)));
typedef float  v8f   __attribute__((ext_vector_type(8)));

union Frag16 { v16bf v; v8bf h[2]; };

// ---------------------------------------------------------------------------
// Async global->LDS staging (CDNA5 ASYNCcnt path), with register fallback.
// Builtin prototype (from hipcc diagnostics): 
//   void __builtin_amdgcn_global_load_async_to_lds_b128(
//       int4 addrspace(1)* src, int4 addrspace(3)* dst, imm offset, imm cpol)
// ---------------------------------------------------------------------------
#if defined(__has_builtin)
#  if __has_builtin(__builtin_amdgcn_global_load_async_to_lds_b128)
#    define USE_ASYNC_LDS 1
#  endif
#endif
#ifndef USE_ASYNC_LDS
#  define USE_ASYNC_LDS 0
#endif

#if USE_ASYNC_LDS
typedef int v4i_t __attribute__((vector_size(16)));
typedef __attribute__((address_space(1))) v4i_t* gbl_v4ip;
typedef __attribute__((address_space(3))) v4i_t* lds_v4ip;
__device__ __forceinline__ gbl_v4ip as_gbl(const bf16_t* p) {
  return (gbl_v4ip)(unsigned long long)p;
}
__device__ __forceinline__ lds_v4ip as_lds(bf16_t* p) {
  // generic LDS pointer: low 32 bits are the LDS byte offset (ISA aperture rule)
  return (lds_v4ip)(unsigned)(unsigned long long)p;
}
__device__ __forceinline__ void wait_async0() {
#  if __has_builtin(__builtin_amdgcn_s_wait_asynccnt)
  __builtin_amdgcn_s_wait_asynccnt(0);
#  else
  asm volatile("s_wait_asynccnt 0x0" ::: "memory");
#  endif
}
#endif

// ---------------------------------------------------------------------------
// fp32 -> bf16 conversion
// ---------------------------------------------------------------------------
__global__ void cvt_f32_bf16(const float* __restrict__ in,
                             bf16_t* __restrict__ out, size_t n) {
  size_t i = (size_t)blockIdx.x * blockDim.x + threadIdx.x;
  if (i < n) out[i] = (bf16_t)in[i];
}

// ---------------------------------------------------------------------------
// bf16 WMMA GEMM:  C[M,N] (f32) = A[M,K] (bf16 row-major) * W[N,K]^T (bf16)
//
// Block: 256 threads = 8 waves, block tile 128x128, K step 32.
// Waves arranged 4x2: wave tile 32x64 -> 8 accumulators (16x16 each).
// A/B tiles double-buffered in LDS; row stride padded to 40 bf16 (20 dwords)
// so 16 consecutive rows hit 16 distinct banks, while ds_load_b128 stays
// 16B-aligned. Next tile staged with async global->LDS loads (ASYNCcnt)
// issued before the compute phase so transfer overlaps ds_load + v_wmma.
// ---------------------------------------------------------------------------
constexpr int LDK = 40;   // padded LDS row stride (elements)

__global__ __launch_bounds__(256)
void gemm_bf16_wmma(const bf16_t* __restrict__ A,
                    const bf16_t* __restrict__ W,
                    float* __restrict__ C,
                    int M, int N, int K) {
  __shared__ bf16_t sA[2][128 * LDK];
  __shared__ bf16_t sB[2][128 * LDK];

  const int tid  = threadIdx.x;
  const int lane = tid & 31;
  const int wave = tid >> 5;
  const int m16  = lane & 15;
  const int hi   = lane >> 4;
  const int wr   = wave & 3;     // wave row (32 rows each)
  const int wc   = wave >> 2;    // wave col (64 cols each)

  const int row0 = blockIdx.x * 128;
  const int col0 = blockIdx.y * 128;

  // cooperative staging: 512 16B chunks per tile, 2 per thread per tile
  const int c0 = tid, c1 = tid + 256;
  const int r0c = c0 >> 2, k0c = (c0 & 3) * 8;
  const int r1c = c1 >> 2, k1c = (c1 & 3) * 8;

  const bf16_t* Ag0 = A + (size_t)(row0 + r0c) * K + k0c;
  const bf16_t* Ag1 = A + (size_t)(row0 + r1c) * K + k1c;
  const bf16_t* Wg0 = W + (size_t)(col0 + r0c) * K + k0c;
  const bf16_t* Wg1 = W + (size_t)(col0 + r1c) * K + k1c;

  v8f acc[2][4];
  #pragma unroll
  for (int s = 0; s < 2; ++s)
    #pragma unroll
    for (int j = 0; j < 4; ++j)
      acc[s][j] = (v8f){0.f,0.f,0.f,0.f,0.f,0.f,0.f,0.f};

#if USE_ASYNC_LDS
  auto stage_async = [&](int kn, int buf) {
    __builtin_amdgcn_global_load_async_to_lds_b128(
        as_gbl(Ag0 + kn), as_lds(&sA[buf][r0c * LDK + k0c]), 0, 0);
    __builtin_amdgcn_global_load_async_to_lds_b128(
        as_gbl(Ag1 + kn), as_lds(&sA[buf][r1c * LDK + k1c]), 0, 0);
    __builtin_amdgcn_global_load_async_to_lds_b128(
        as_gbl(Wg0 + kn), as_lds(&sB[buf][r0c * LDK + k0c]), 0, 0);
    __builtin_amdgcn_global_load_async_to_lds_b128(
        as_gbl(Wg1 + kn), as_lds(&sB[buf][r1c * LDK + k1c]), 0, 0);
  };
  stage_async(0, 0);
  wait_async0();
  __syncthreads();
#else
  v8bf ra0 = *(const v8bf*)(Ag0);
  v8bf ra1 = *(const v8bf*)(Ag1);
  v8bf rb0 = *(const v8bf*)(Wg0);
  v8bf rb1 = *(const v8bf*)(Wg1);
  *(v8bf*)&sA[0][r0c * LDK + k0c] = ra0;
  *(v8bf*)&sA[0][r1c * LDK + k1c] = ra1;
  *(v8bf*)&sB[0][r0c * LDK + k0c] = rb0;
  *(v8bf*)&sB[0][r1c * LDK + k1c] = rb1;
  __syncthreads();
#endif

  const int steps = K / 32;
  for (int it = 0; it < steps; ++it) {
    const int buf = it & 1;
    const bool more = (it + 1) < steps;

    // stage next tile early (overlaps with ds + wmma below); buf^1 is not
    // read by anyone this iteration (its last readers finished before the
    // barrier that ended the previous iteration)
    if (more) {
      const int kn = (it + 1) * 32;
#if USE_ASYNC_LDS
      stage_async(kn, buf ^ 1);
#else
      ra0 = *(const v8bf*)(Ag0 + kn);
      ra1 = *(const v8bf*)(Ag1 + kn);
      rb0 = *(const v8bf*)(Wg0 + kn);
      rb1 = *(const v8bf*)(Wg1 + kn);
#endif
      __builtin_prefetch(Ag0 + kn + 64, 0, 3);
      __builtin_prefetch(Wg0 + kn + 64, 0, 3);
    }

    // fragments from LDS (distinct registers -> wmma chain without stalls)
    Frag16 af[2], bfr[4];
    #pragma unroll
    for (int s = 0; s < 2; ++s) {
      const int r = (wr * 32 + s * 16 + m16) * LDK + hi * 8;
      af[s].h[0] = *(const v8bf*)&sA[buf][r];
      af[s].h[1] = *(const v8bf*)&sA[buf][r + 16];
    }
    #pragma unroll
    for (int j = 0; j < 4; ++j) {
      const int r = (wc * 64 + j * 16 + m16) * LDK + hi * 8;
      bfr[j].h[0] = *(const v8bf*)&sB[buf][r];
      bfr[j].h[1] = *(const v8bf*)&sB[buf][r + 16];
    }

    #pragma unroll
    for (int s = 0; s < 2; ++s)
      #pragma unroll
      for (int j = 0; j < 4; ++j)
        acc[s][j] = __builtin_amdgcn_wmma_f32_16x16x32_bf16(
            false, af[s].v, false, bfr[j].v, (short)0, acc[s][j], false, false);

    if (more) {
#if USE_ASYNC_LDS
      wait_async0();                           // own async writes done
      __syncthreads();                         // all waves' writes visible
#else
      __syncthreads();                         // everyone done reading buf^1
      *(v8bf*)&sA[buf ^ 1][r0c * LDK + k0c] = ra0;
      *(v8bf*)&sA[buf ^ 1][r1c * LDK + k1c] = ra1;
      *(v8bf*)&sB[buf ^ 1][r0c * LDK + k0c] = rb0;
      *(v8bf*)&sB[buf ^ 1][r1c * LDK + k1c] = rb1;
      __syncthreads();                         // next tile visible
#endif
    }
  }

  // C/D layout: VGPR r -> M = tile_row + hi*8 + r ; N = tile_col + m16
  #pragma unroll
  for (int s = 0; s < 2; ++s) {
    #pragma unroll
    for (int j = 0; j < 4; ++j) {
      const int row = row0 + wr * 32 + s * 16 + hi * 8;
      const int col = col0 + wc * 64 + j * 16 + m16;
      #pragma unroll
      for (int r = 0; r < 8; ++r)
        C[(size_t)(row + r) * N + col] = acc[s][j][r];
    }
  }
}

// ---------------------------------------------------------------------------
// ba projection + gates: beta = sigmoid(b), g = -exp(A_log)*softplus(a+dt_bias)
// one block per (b,t), 64 threads (one per ba output)
// ---------------------------------------------------------------------------
__global__ __launch_bounds__(64)
void ba_gates_kernel(const float* __restrict__ hs,
                     const float* __restrict__ W_ba,
                     const float* __restrict__ dt_bias,
                     const float* __restrict__ A_log,
                     float* __restrict__ beta,
                     float* __restrict__ g) {
  const int bt = blockIdx.x;
  const int o  = threadIdx.x;          // 0..63
  const float4* x4 = (const float4*)(hs + (size_t)bt * H_);
  const float4* w4 = (const float4*)(W_ba + (size_t)o * H_);
  float s = 0.f;
  for (int i = 0; i < H_ / 4; ++i) {
    float4 a = x4[i], w = w4[i];
    s += a.x * w.x + a.y * w.y + a.z * w.z + a.w * w.w;
  }
  const int kh = o >> 2, j = o & 3;
  if (j < 2) {                          // b -> beta
    const int vh = kh * R_ + j;
    beta[(size_t)bt * NUM_V + vh] = 1.f / (1.f + expf(-s));
  } else {                              // a -> g
    const int vh = kh * R_ + (j - 2);
    float x = s + dt_bias[vh];
    float sp = (x > 20.f) ? x : log1pf(expf(x));
    g[(size_t)bt * NUM_V + vh] = -expf(A_log[vh]) * sp;
  }
}

// ---------------------------------------------------------------------------
// depthwise causal conv (K=4) + SiLU, with the key-head-grouped reordering
// of qkvz folded into the gather. Writes q/k/v buffers.
// ---------------------------------------------------------------------------
__global__ void conv_silu_kernel(const float* __restrict__ qkvz,
                                 const float* __restrict__ conv_w,
                                 float* __restrict__ qb,
                                 float* __restrict__ kb,
                                 float* __restrict__ vb) {
  size_t idx = (size_t)blockIdx.x * blockDim.x + threadIdx.x;
  if (idx >= (size_t)BT * CONV_DIM) return;
  const int c  = (int)(idx % CONV_DIM);
  const int bt = (int)(idx / CONV_DIM);
  const int t  = bt % T_;

  int inner; float* outp; int oidx;
  if (c < KEY_DIM) {                       // q
    const int kh = c >> 7, d = c & 127;
    inner = kh * GRP + d;
    outp = qb; oidx = bt * KEY_DIM + c;
  } else if (c < 2 * KEY_DIM) {            // k
    const int cc = c - KEY_DIM, kh = cc >> 7, d = cc & 127;
    inner = kh * GRP + DK + d;
    outp = kb; oidx = bt * KEY_DIM + cc;
  } else {                                 // v
    const int cc = c - 2 * KEY_DIM, vh = cc >> 7, d = cc & 127;
    const int kh = vh >> 1, r = vh & 1;
    inner = kh * GRP + 2 * DK + r * DV + d;
    outp = vb; oidx = bt * VAL_DIM + cc;
  }

  float acc = 0.f;
  #pragma unroll
  for (int j = 0; j < KCONV; ++j) {
    const int tt = t - (KCONV - 1) + j;
    if (tt >= 0)
      acc += qkvz[((size_t)(bt - t + tt)) * QKVZ_DIM + inner] * conv_w[c * KCONV + j];
  }
  outp[(size_t)oidx] = acc / (1.f + expf(-acc));   // SiLU
}

// ---------------------------------------------------------------------------
// Gated delta-rule recurrence. One workgroup per (b, value-head).
// 128 threads; thread v owns state column S[:,v] in 128 VGPRs.
// k_t / q_t broadcast through LDS (conflict-free same-address reads).
// t+1 operands prefetched into registers so global latency hides under
// the 3x128 FMA chain of step t.
// ---------------------------------------------------------------------------
__global__ __launch_bounds__(128)
void delta_rule_kernel(const float* __restrict__ qb,
                       const float* __restrict__ kb,
                       const float* __restrict__ vb,
                       const float* __restrict__ beta,
                       const float* __restrict__ g,
                       float* __restrict__ core) {
  const int bh = blockIdx.x;          // 0..63
  const int bb = bh >> 5;
  const int vh = bh & 31;
  const int kh = vh >> 1;             // GQA: repeat q,k over 2 value heads
  const int v  = threadIdx.x;         // 0..127

  float S[DK];
  #pragma unroll
  for (int k = 0; k < DK; ++k) S[k] = 0.f;

  __shared__ float kq[2 * DK];        // normalized k_t then q_t
  __shared__ float pq[4], pk[4];

  constexpr float qscale = 0.08838834764831845f;   // DK^-0.5

  const size_t qk_base = (size_t)bb * T_ * KEY_DIM + kh * DK + v;
  const size_t v_base  = (size_t)bb * T_ * VAL_DIM + vh * DV + v;
  const size_t g_base  = (size_t)bb * T_ * NUM_V + vh;

  // preload t = 0
  float qd = qb[qk_base];
  float kd = kb[qk_base];
  float vt = vb[v_base];
  float gt  = g[g_base];
  float bt_ = beta[g_base];

  for (int t = 0; t < T_; ++t) {
    // issue t+1 loads early
    float qd_n = 0.f, kd_n = 0.f, vt_n = 0.f, gt_n = 0.f, bt_n = 0.f;
    if (t + 1 < T_) {
      qd_n = qb[qk_base + (size_t)(t + 1) * KEY_DIM];
      kd_n = kb[qk_base + (size_t)(t + 1) * KEY_DIM];
      vt_n = vb[v_base  + (size_t)(t + 1) * VAL_DIM];
      gt_n = g[g_base   + (size_t)(t + 1) * NUM_V];
      bt_n = beta[g_base + (size_t)(t + 1) * NUM_V];
    }

    // l2-norm reductions over the 128-dim head
    float sq = qd * qd, sk = kd * kd;
    #pragma unroll
    for (int off = 16; off > 0; off >>= 1) {
      sq += __shfl_xor(sq, off);
      sk += __shfl_xor(sk, off);
    }
    if ((threadIdx.x & 31) == 0) { pq[threadIdx.x >> 5] = sq; pk[threadIdx.x >> 5] = sk; }
    __syncthreads();
    sq = pq[0] + pq[1] + pq[2] + pq[3];
    sk = pk[0] + pk[1] + pk[2] + pk[3];

    kq[v]      = kd * rsqrtf(sk + 1e-6f);
    kq[DK + v] = qd * rsqrtf(sq + 1e-6f) * qscale;
    __syncthreads();

    const float eg = __expf(gt);

    // kv = eg * (S_old^T k)
    float kv = 0.f;
    #pragma unroll
    for (int k = 0; k < DK; ++k) kv += S[k] * kq[k];
    kv *= eg;
    const float delta = (vt - kv) * bt_;

    // S = S*eg + k (outer) delta ; o = S_new^T q
    float o = 0.f;
    #pragma unroll
    for (int k = 0; k < DK; ++k) {
      S[k] = S[k] * eg + kq[k] * delta;
      o += S[k] * kq[DK + k];
    }
    core[(size_t)(bb * T_ + t) * VAL_DIM + vh * DV + v] = o;
    __syncthreads();   // protect kq/pq/pk before next step

    qd = qd_n; kd = kd_n; vt = vt_n; gt = gt_n; bt_ = bt_n;
  }
}

// ---------------------------------------------------------------------------
// FusedRMSNormGated: rmsnorm(core) * w * silu(z), emit bf16 for final GEMM.
// One block per (b,t,vh); 128 threads.
// ---------------------------------------------------------------------------
__global__ __launch_bounds__(128)
void gated_norm_kernel(const float* __restrict__ core,
                       const float* __restrict__ qkvz,
                       const float* __restrict__ norm_w,
                       bf16_t* __restrict__ out_bf16) {
  const int grp = blockIdx.x;          // 0 .. BT*NUM_V-1
  const int d   = threadIdx.x;
  const int vh  = grp & 31;
  const size_t bt = (size_t)(grp >> 5);
  const int kh = vh >> 1, r = vh & 1;

  const float x = core[(size_t)grp * DV + d];
  const float z = qkvz[bt * QKVZ_DIM + kh * GRP + 2 * DK + R_ * DV + r * DV + d];

  float s = x * x;
  #pragma unroll
  for (int off = 16; off > 0; off >>= 1) s += __shfl_xor(s, off);
  __shared__ float part[4];
  if ((threadIdx.x & 31) == 0) part[threadIdx.x >> 5] = s;
  __syncthreads();
  s = part[0] + part[1] + part[2] + part[3];

  const float inv = rsqrtf(s * (1.f / DV) + 1e-6f);
  const float sz  = z / (1.f + expf(-z));          // silu(z)
  out_bf16[(size_t)grp * DV + d] = (bf16_t)(x * inv * norm_w[d] * sz);
}

// ---------------------------------------------------------------------------
// Launch
// ---------------------------------------------------------------------------
extern "C" void kernel_launch(void* const* d_in, const int* in_sizes, int n_in,
                              void* d_out, int out_size, void* d_ws, size_t ws_size,
                              hipStream_t stream) {
  (void)in_sizes; (void)n_in; (void)out_size; (void)ws_size;
  const float* hidden   = (const float*)d_in[0];   // [B,T,H]
  const float* W_qkvz   = (const float*)d_in[1];   // [12288, H]
  const float* W_ba     = (const float*)d_in[2];   // [64, H]
  const float* conv_w   = (const float*)d_in[3];   // [8192, 4]
  const float* dt_bias  = (const float*)d_in[4];   // [32]
  const float* A_log    = (const float*)d_in[5];   // [32]
  const float* norm_w   = (const float*)d_in[6];   // [128]
  const float* W_out    = (const float*)d_in[7];   // [H, 4096]
  float* out = (float*)d_out;                      // [B,T,H] fp32

  // workspace carve-out
  char* ws = (char*)d_ws;
  size_t off = 0;
  auto carve = [&](size_t bytes) -> char* {
    char* p = ws + off;
    off += (bytes + 255) & ~(size_t)255;
    return p;
  };
  bf16_t* hid_bf   = (bf16_t*)carve((size_t)BT * H_ * 2);
  bf16_t* wqkvz_bf = (bf16_t*)carve((size_t)QKVZ_DIM * H_ * 2);
  bf16_t* wout_bf  = (bf16_t*)carve((size_t)H_ * VAL_DIM * 2);
  float*  qkvz     = (float*)carve((size_t)BT * QKVZ_DIM * 4);
  float*  qbuf     = (float*)carve((size_t)BT * KEY_DIM * 4);
  float*  kbuf     = (float*)carve((size_t)BT * KEY_DIM * 4);
  float*  vbuf     = (float*)carve((size_t)BT * VAL_DIM * 4);
  float*  betab    = (float*)carve((size_t)BT * NUM_V * 4);
  float*  gbuf     = (float*)carve((size_t)BT * NUM_V * 4);
  float*  coreb    = (float*)carve((size_t)BT * VAL_DIM * 4);
  bf16_t* core_bf  = (bf16_t*)carve((size_t)BT * VAL_DIM * 2);

  auto cvt = [&](const float* src, bf16_t* dst, size_t n) {
    cvt_f32_bf16<<<dim3((unsigned)((n + 255) / 256)), dim3(256), 0, stream>>>(src, dst, n);
  };
  cvt(hidden, hid_bf,   (size_t)BT * H_);
  cvt(W_qkvz, wqkvz_bf, (size_t)QKVZ_DIM * H_);
  cvt(W_out,  wout_bf,  (size_t)H_ * VAL_DIM);

  // qkvz projection: [4096,2048] x [2048,12288]
  gemm_bf16_wmma<<<dim3(BT / 128, QKVZ_DIM / 128), dim3(256), 0, stream>>>(
      hid_bf, wqkvz_bf, qkvz, BT, QKVZ_DIM, H_);

  // ba projection + gate computation
  ba_gates_kernel<<<dim3(BT), dim3(64), 0, stream>>>(
      hidden, W_ba, dt_bias, A_log, betab, gbuf);

  // depthwise conv + SiLU (with reordering gather)
  {
    size_t n = (size_t)BT * CONV_DIM;
    conv_silu_kernel<<<dim3((unsigned)((n + 255) / 256)), dim3(256), 0, stream>>>(
        qkvz, conv_w, qbuf, kbuf, vbuf);
  }

  // gated delta-rule recurrence: 64 heads, one workgroup each
  delta_rule_kernel<<<dim3(B_ * NUM_V), dim3(128), 0, stream>>>(
      qbuf, kbuf, vbuf, betab, gbuf, coreb);

  // gated RMSNorm -> bf16
  gated_norm_kernel<<<dim3(BT * NUM_V), dim3(128), 0, stream>>>(
      coreb, qkvz, norm_w, core_bf);

  // output projection: [4096,4096] x [4096,2048] -> d_out (fp32)
  gemm_bf16_wmma<<<dim3(BT / 128, H_ / 128), dim3(256), 0, stream>>>(
      core_bf, wout_bf, out, BT, H_, VAL_DIM);
}
